// GNN_63556926046385
// MI455X (gfx1250) — matrix operand order repaired and checked by reading, not compiled, not run on hardware
//
#include <hip/hip_runtime.h>

// ---------------------------------------------------------------------------
// 2-layer GCN for MI455X (gfx1250, wave32).
//   GEMMs:      v_wmma_f32_16x16x32_bf16 (bf16 in, f32 accumulate),
//               A-loads batched per row-tile so the 4 WMMAs issue back-to-back
//   Aggregation: wave-per-edge coalesced global_atomic_add_f32 (L2 resident)
// ---------------------------------------------------------------------------

typedef __attribute__((ext_vector_type(16))) __bf16 v16bf;
typedef __attribute__((ext_vector_type(8)))  __bf16 v8bf;
typedef __attribute__((ext_vector_type(8)))  float  v8f;

#define FEAT 128   // IN_C == HID == 128

// ---------------- elementwise helpers ----------------

__global__ __launch_bounds__(256) void k_zero_f32(float* __restrict__ p, int n) {
  int i = blockIdx.x * 256 + threadIdx.x;
  if (i < n) p[i] = 0.0f;
}

__global__ __launch_bounds__(256) void k_degree(const long long* __restrict__ ei,
                                                float* __restrict__ deg, int E) {
  int e = blockIdx.x * 256 + threadIdx.x;
  if (e < E) {
    int d = (int)ei[(size_t)E + e];   // dst row
    unsafeAtomicAdd(&deg[d], 1.0f);
  }
}

__global__ __launch_bounds__(256) void k_rsqrt(float* __restrict__ dinv, int n) {
  int i = blockIdx.x * 256 + threadIdx.x;
  if (i < n) dinv[i] = __frsqrt_rn(dinv[i] + 1.0f);   // +1 self-loop; deg>=1 always
}

__global__ __launch_bounds__(256) void k_cast_bf16(const float* __restrict__ x,
                                                   __bf16* __restrict__ a, int total) {
  int i = blockIdx.x * 256 + threadIdx.x;
  if (i < total) a[i] = (__bf16)x[i];
}

// out[n,c] = dinv[n]^2 * H[n,c]   (self-loop term initializes the accumulator)
__global__ __launch_bounds__(256) void k_selfloop(const float* __restrict__ H,
                                                  const float* __restrict__ dinv,
                                                  float* __restrict__ O, int total) {
  int i = blockIdx.x * 256 + threadIdx.x;
  if (i < total) {
    float dv = dinv[i >> 7];
    O[i] = dv * dv * H[i];
  }
}

// Abf[n,c] = bf16(relu(H[n,c] + b[c]))
__global__ __launch_bounds__(256) void k_bias_relu_cast(const float* __restrict__ H,
                                                        const float* __restrict__ b,
                                                        __bf16* __restrict__ a, int total) {
  int i = blockIdx.x * 256 + threadIdx.x;
  if (i < total) {
    float v = H[i] + b[i & (FEAT - 1)];
    a[i] = (__bf16)fmaxf(v, 0.0f);
  }
}

// ---------------- WMMA GEMM: C[N,128] = A[N,128] @ W[128,128] ----------------
// Block: 256 threads = 8 waves. Wave w owns column tile w (16 cols).
// Each block processes 8 row-tiles (128 rows); W staged in LDS as bf16.
// Per row-tile: issue all 8 A global_load_b128 first (one clause, one wait),
// then run the 4 WMMAs back-to-back on the accumulation chain.

__global__ __launch_bounds__(256) void k_gemm_bf16(const __bf16* __restrict__ A,
                                                   const float*  __restrict__ Wf,
                                                   float* __restrict__ C,
                                                   int nRowTiles) {
  __shared__ __bf16 Ws[FEAT * FEAT];           // 32 KB
  for (int i = threadIdx.x; i < FEAT * FEAT; i += 256)
    Ws[i] = (__bf16)Wf[i];
  __syncthreads();

  const int lane    = threadIdx.x & 31;
  const int colTile = threadIdx.x >> 5;        // 0..7
  const int nB      = (lane & 15) + colTile * 16;
  const int kbB     = (lane >> 4) * 8;         // B: two 8-elem chunks at kbB, kbB+16

  // B fragments for the 4 K-tiles (K = kt*32 + kbB + i + (i & 8)), column nB
  v16bf Bfrag[4];
#pragma unroll
  for (int kt = 0; kt < 4; ++kt) {
#pragma unroll
    for (int i = 0; i < 16; ++i)
      Bfrag[kt][i] = Ws[(kt * 32 + kbB + i + (i & 8)) * FEAT + nB];
  }

  const int mA  = lane & 15;                   // A row within tile
  const int kbA = (lane >> 4) * 8;             // A: two 8-elem chunks at kbA, kbA+16

  const int rtBase = blockIdx.x * 8;
#pragma unroll 1
  for (int rr = 0; rr < 8; ++rr) {
    int rt = rtBase + rr;
    if (rt >= nRowTiles) break;
    const __bf16* Ab = A + (size_t)rt * 16 * FEAT + (size_t)mA * FEAT + kbA;

    // ---- batch all 8 independent 16B loads for this row-tile ----
    v8bf ch[8];
#pragma unroll
    for (int kt = 0; kt < 4; ++kt) {
      const __bf16* ap = Ab + kt * 32;
      ch[2 * kt + 0] = *(const v8bf*)(ap);        // K = base .. base+7
      ch[2 * kt + 1] = *(const v8bf*)(ap + 16);   // K = base+16 .. base+23
    }

    // ---- 4 back-to-back WMMAs on the accumulation chain ----
    v8f acc = {};
#pragma unroll
    for (int kt = 0; kt < 4; ++kt) {
      v16bf Af;
#pragma unroll
      for (int i = 0; i < 8; ++i) {
        Af[i]     = ch[2 * kt + 0][i];
        Af[8 + i] = ch[2 * kt + 1][i];
      }
      acc = __builtin_amdgcn_wmma_f32_16x16x32_bf16(
          false, Af, false, Bfrag[kt], (short)0, acc, false, false);
    }

    // C/D layout: VGPR r -> row r + 8*(lane>>4), col lane&15
    float* Cp = C + (size_t)rt * 16 * FEAT + colTile * 16 + (lane & 15);
    const int rofs = (lane >> 4) * 8;
#pragma unroll
    for (int r = 0; r < 8; ++r)
      Cp[(size_t)(r + rofs) * FEAT] = acc[r];
  }
}

// ---------------- edge scatter: O[dst] += dinv[src]*dinv[dst]*H[src] --------
// One wave per edge: lanes cover 128 channels via float4 -> 512B coalesced
// atomic burst per edge, L2-resident.

__global__ __launch_bounds__(256) void k_scatter(const long long* __restrict__ ei,
                                                 const float* __restrict__ dinv,
                                                 const float* __restrict__ H,
                                                 float* __restrict__ O, int E) {
  const int lane = threadIdx.x & 31;
  const int wid  = (blockIdx.x * 256 + threadIdx.x) >> 5;
  const int nw   = (gridDim.x * 256) >> 5;
  for (int e = wid; e < E; e += nw) {
    int s = (int)ei[e];
    int d = (int)ei[(size_t)E + e];
    float nrm = dinv[s] * dinv[d];
    float4 v = *((const float4*)(H + (size_t)s * FEAT) + lane);
    float* op = O + (size_t)d * FEAT + lane * 4;
    unsafeAtomicAdd(op + 0, v.x * nrm);
    unsafeAtomicAdd(op + 1, v.y * nrm);
    unsafeAtomicAdd(op + 2, v.z * nrm);
    unsafeAtomicAdd(op + 3, v.w * nrm);
  }
}

// ---------------- final: out[n] = bl + sum_c relu(H[n,c]+b2[c]) * Wl[c] -----

__global__ __launch_bounds__(256) void k_final(const float* __restrict__ H,
                                               const float* __restrict__ b2,
                                               const float* __restrict__ Wl,
                                               const float* __restrict__ bl,
                                               float* __restrict__ out, int N) {
  const int lane = threadIdx.x & 31;
  const int wid  = (blockIdx.x * 256 + threadIdx.x) >> 5;
  const int nw   = (gridDim.x * 256) >> 5;
  for (int n = wid; n < N; n += nw) {
    float4 v  = *((const float4*)(H + (size_t)n * FEAT) + lane);
    float4 bb = ((const float4*)b2)[lane];
    float4 w  = ((const float4*)Wl)[lane];
    float s = fmaxf(v.x + bb.x, 0.0f) * w.x
            + fmaxf(v.y + bb.y, 0.0f) * w.y
            + fmaxf(v.z + bb.z, 0.0f) * w.z
            + fmaxf(v.w + bb.w, 0.0f) * w.w;
#pragma unroll
    for (int off = 16; off > 0; off >>= 1)
      s += __shfl_down(s, off, 32);
    if (lane == 0) out[n] = s + bl[0];
  }
}

// ---------------------------------------------------------------------------

extern "C" void kernel_launch(void* const* d_in, const int* in_sizes, int n_in,
                              void* d_out, int out_size, void* d_ws, size_t ws_size,
                              hipStream_t stream) {
  const float*     x  = (const float*)d_in[0];
  const long long* ei = (const long long*)d_in[1];   // int64 edge_index [2,E]
  const float*     W1 = (const float*)d_in[2];
  const float*     b1 = (const float*)d_in[3];
  const float*     W2 = (const float*)d_in[4];
  const float*     b2 = (const float*)d_in[5];
  const float*     Wl = (const float*)d_in[6];
  const float*     bl = (const float*)d_in[7];
  float* out = (float*)d_out;

  const int N = in_sizes[0] / FEAT;     // 100000
  const int E = in_sizes[1] / 2;        // 1600000
  const int total = N * FEAT;
  const int nRowTiles = (N + 15) / 16;

  // workspace layout
  char* ws = (char*)d_ws;
  float*  dinv = (float*)ws;                                   //  N f32   (<1 MB)
  __bf16* Abf  = (__bf16*)(ws + ((size_t)1  << 20));           //  N*128 bf16 (25.6 MB)
  float*  Hlin = (float*)(ws + ((size_t)29 << 20));            //  N*128 f32  (51.2 MB)
  float*  Hagg = (float*)(ws + ((size_t)81 << 20));            //  N*128 f32  (51.2 MB)

  const int eb  = (E + 255) / 256;
  const int nb  = (N + 255) / 256;
  const int tb  = (total + 255) / 256;
  const int gb  = (nRowTiles + 7) / 8;
  const int sb  = 4096;                  // scatter blocks (32768 waves)
  const int fb  = (N + 7) / 8;           // final: 8 waves/block, wave per node

  // normalization: deg (with self-loop) -> deg^{-1/2}
  k_zero_f32<<<nb, 256, 0, stream>>>(dinv, N);
  k_degree  <<<eb, 256, 0, stream>>>(ei, dinv, E);
  k_rsqrt   <<<nb, 256, 0, stream>>>(dinv, N);

  // layer 1
  k_cast_bf16<<<tb, 256, 0, stream>>>(x, Abf, total);
  k_gemm_bf16<<<gb, 256, 0, stream>>>(Abf, W1, Hlin, nRowTiles);
  k_selfloop <<<tb, 256, 0, stream>>>(Hlin, dinv, Hagg, total);
  k_scatter  <<<sb, 256, 0, stream>>>(ei, dinv, Hlin, Hagg, E);
  k_bias_relu_cast<<<tb, 256, 0, stream>>>(Hagg, b1, Abf, total);

  // layer 2
  k_gemm_bf16<<<gb, 256, 0, stream>>>(Abf, W2, Hlin, nRowTiles);
  k_selfloop <<<tb, 256, 0, stream>>>(Hlin, dinv, Hagg, total);
  k_scatter  <<<sb, 256, 0, stream>>>(ei, dinv, Hlin, Hagg, E);

  // fused relu + linear head
  k_final<<<fb, 256, 0, stream>>>(Hagg, b2, Wl, bl, out, N);
}
